// GTN_high_40132174414154
// MI455X (gfx1250) — matrix operand order repaired and checked by reading, not compiled, not run on hardware
//
#include <hip/hip_runtime.h>
#include <hip/hip_bf16.h>

typedef __attribute__((ext_vector_type(16))) _Float16 v16h;
typedef __attribute__((ext_vector_type(8)))  _Float16 v8h;
typedef __attribute__((ext_vector_type(4)))  _Float16 v4h;
typedef __attribute__((ext_vector_type(8)))  float    v8f;

#define LDSTR 40  // LDS row stride in halves: 80B, 16B-aligned, bank-spread
#define AS3 __attribute__((address_space(3)))

static __device__ __forceinline__ v16h pack16h(v8h a, v8h b) {
  v16h r;
#pragma unroll
  for (int i = 0; i < 8; ++i) { r[i] = a[i]; r[i + 8] = b[i]; }
  return r;
}

static __device__ __forceinline__ v16h ld16h(const _Float16* p) {
  return pack16h(*(const v8h*)p, *(const v8h*)(p + 8));
}

static __device__ __forceinline__ v8f wmma_f16(v16h a, v16h b, v8f c) {
  return __builtin_amdgcn_wmma_f32_16x16x32_f16(false, a, false, b, (short)0, c,
                                                false, false);
}

// ---------------------------------------------------------------------------
// Generic tiled GEMM: C[M,Ncol] = act(A[M,K] @ B[K,Ncol] + bias)
// fp32 in/out, f16 WMMA with fp32 accumulate. BM=BN=64, BK=32, 4 waves.
// ---------------------------------------------------------------------------
__global__ __launch_bounds__(128, 1) void gemm_kernel(
    const float* __restrict__ A, const float* __restrict__ B,
    const float* __restrict__ bias, float* __restrict__ C,
    int M, int K, int Ncol, int act)
{
  __shared__ __align__(16) _Float16 As[64 * LDSTR];
  __shared__ __align__(16) _Float16 Bt[64 * LDSTR];  // transposed: [col][k]
  const int tid  = threadIdx.x;
  const int wave = tid >> 5;
  const int lane = tid & 31;
  const int lh   = lane >> 4;
  const int lm   = lane & 15;
  const int rowBase = blockIdx.y * 64;
  const int colBase = blockIdx.x * 64;

  v8f acc[4] = {};

  for (int k0 = 0; k0 < K; k0 += 32) {
#pragma unroll
    for (int i = 0; i < 4; ++i) {           // A tile 64x32 -> f16 LDS (packed)
      int s  = tid + i * 128;
      int r  = s >> 3;
      int c4 = (s & 7) * 4;
      float4 v = *(const float4*)(A + (size_t)(rowBase + r) * K + k0 + c4);
      v4h hv = { (_Float16)v.x, (_Float16)v.y, (_Float16)v.z, (_Float16)v.w };
      *(v4h*)&As[r * LDSTR + c4] = hv;
    }
#pragma unroll
    for (int i = 0; i < 4; ++i) {           // B tile 32x64 -> transposed f16 LDS
      int s  = tid + i * 128;
      int kr = s >> 4;
      int c4 = (s & 15) * 4;
      float4 v = *(const float4*)(B + (size_t)(k0 + kr) * Ncol + colBase + c4);
      Bt[(c4 + 0) * LDSTR + kr] = (_Float16)v.x;
      Bt[(c4 + 1) * LDSTR + kr] = (_Float16)v.y;
      Bt[(c4 + 2) * LDSTR + kr] = (_Float16)v.z;
      Bt[(c4 + 3) * LDSTR + kr] = (_Float16)v.w;
    }
    __syncthreads();

    // A fragment for this wave's 16 rows (contiguous k = 16*lh + s, same order as the B fragment)
    const _Float16* arow = &As[(wave * 16 + lm) * LDSTR];
    v16h af = pack16h(*(const v8h*)(arow + lh * 16),
                      *(const v8h*)(arow + lh * 16 + 8));
#pragma unroll
    for (int ct = 0; ct < 4; ++ct) {
      v16h bf = ld16h(&Bt[(ct * 16 + lm) * LDSTR + lh * 16]);
      acc[ct] = wmma_f16(af, bf, acc[ct]);
    }
    __syncthreads();
  }

#pragma unroll
  for (int ct = 0; ct < 4; ++ct) {
    int col = colBase + ct * 16 + lm;
    float bv = bias ? bias[col] : 0.f;
#pragma unroll
    for (int r = 0; r < 8; ++r) {
      int row = rowBase + wave * 16 + lh * 8 + r;
      float v = acc[ct][r] + bv;
      if (act == 1) v = fmaxf(v, 0.f);
      C[(size_t)row * Ncol + col] = v;
    }
  }
}

// ---------------------------------------------------------------------------
// Flash attention: one wave per (16-row Q tile, head). DH=32 => K-dim of one
// v_wmma_f32_16x16x32_f16. Streams over 32-key blocks with online softmax.
// qkv: [N,768] (q|k|v), out: [N,256]
// ---------------------------------------------------------------------------
__global__ __launch_bounds__(32, 1) void flash_kernel(
    const float* __restrict__ qkv, float* __restrict__ outp)
{
  constexpr int D3 = 768, Dm = 256;
  const float scale = 0.17677669529663687f;  // 1/sqrt(32)
  const int lane = threadIdx.x;
  const int lh = lane >> 4, lm = lane & 15;
  const int q0 = blockIdx.x * 16;
  const int hh = blockIdx.y;

  __shared__ __align__(16) _Float16 Ks[32 * LDSTR];  // [key][d]
  __shared__ __align__(16) _Float16 Vt[32 * LDSTR];  // [d][key]
  __shared__ __align__(16) _Float16 Ps[16 * LDSTR];  // P re-layout buffer

  const float* qp = qkv + (size_t)(q0 + lm) * D3 + hh * 32;
  v16h qf;
#pragma unroll
  for (int j = 0; j < 8; ++j) {
    qf[j]     = (_Float16)(qp[lh * 16 + j] * scale);
    qf[j + 8] = (_Float16)(qp[lh * 16 + 8 + j] * scale);
  }

  float mrow[8], lrow[8];
#pragma unroll
  for (int r = 0; r < 8; ++r) { mrow[r] = -3.0e38f; lrow[r] = 0.f; }
  v8f o0 = {}, o1 = {};

  for (int kb = 0; kb < 4096; kb += 32) {
    const float* kp = qkv + (size_t)(kb + lane) * D3 + Dm + hh * 32;
    const float* vp = qkv + (size_t)(kb + lane) * D3 + 2 * Dm + hh * 32;
    if (kb + 32 < 4096) {                       // prefetch next key/value block
      __builtin_prefetch(kp + 32 * D3, 0, 1);
      __builtin_prefetch(vp + 32 * D3, 0, 1);
    }
#pragma unroll
    for (int d4 = 0; d4 < 32; d4 += 4) {
      float4 kv = *(const float4*)(kp + d4);
      float4 vv = *(const float4*)(vp + d4);
      v4h kh = { (_Float16)kv.x, (_Float16)kv.y, (_Float16)kv.z, (_Float16)kv.w };
      *(v4h*)&Ks[lane * LDSTR + d4] = kh;
      Vt[(d4 + 0) * LDSTR + lane] = (_Float16)vv.x;
      Vt[(d4 + 1) * LDSTR + lane] = (_Float16)vv.y;
      Vt[(d4 + 2) * LDSTR + lane] = (_Float16)vv.z;
      Vt[(d4 + 3) * LDSTR + lane] = (_Float16)vv.w;
    }
    __syncthreads();

    v16h kf0 = ld16h(&Ks[lm * LDSTR + lh * 16]);
    v16h kf1 = ld16h(&Ks[(lm + 16) * LDSTR + lh * 16]);
    v8f s0 = {}, s1 = {};
    s0 = wmma_f16(qf, kf0, s0);
    s1 = wmma_f16(qf, kf1, s1);

#pragma unroll
    for (int r = 0; r < 8; ++r) {
      float mx = fmaxf(s0[r], s1[r]);
#pragma unroll
      for (int sft = 1; sft < 16; sft <<= 1) mx = fmaxf(mx, __shfl_xor(mx, sft, 16));
      float mnew = fmaxf(mrow[r], mx);
      float corr = __expf(mrow[r] - mnew);
      float e0 = __expf(s0[r] - mnew);
      float e1 = __expf(s1[r] - mnew);
      float rs = e0 + e1;
#pragma unroll
      for (int sft = 1; sft < 16; sft <<= 1) rs += __shfl_xor(rs, sft, 16);
      lrow[r] = lrow[r] * corr + rs;
      mrow[r] = mnew;
      o0[r] *= corr; o1[r] *= corr;
      Ps[(lh * 8 + r) * LDSTR + lm]      = (_Float16)e0;
      Ps[(lh * 8 + r) * LDSTR + lm + 16] = (_Float16)e1;
    }
    __syncthreads();

    const _Float16* prow = &Ps[lm * LDSTR];
    v16h pf = pack16h(*(const v8h*)(prow + lh * 16),
                      *(const v8h*)(prow + lh * 16 + 8));
    v16h vf0 = ld16h(&Vt[lm * LDSTR + lh * 16]);
    v16h vf1 = ld16h(&Vt[(lm + 16) * LDSTR + lh * 16]);
    o0 = wmma_f16(pf, vf0, o0);
    o1 = wmma_f16(pf, vf1, o1);
    __syncthreads();
  }

#pragma unroll
  for (int r = 0; r < 8; ++r) {
    float inv = 1.f / lrow[r];
    int row = q0 + lh * 8 + r;
    float* op = outp + (size_t)row * Dm + hh * 32;
    op[lm]      = o0[r] * inv;
    op[lm + 16] = o1[r] * inv;
  }
}

// ---------------------------------------------------------------------------
// Residual + LayerNorm: out = LN(x + d) * g + b. One wave per 256-wide row.
// ---------------------------------------------------------------------------
__global__ __launch_bounds__(256, 1) void ln_res_kernel(
    const float* __restrict__ x, const float* __restrict__ dlt,
    const float* __restrict__ g, const float* __restrict__ b,
    float* __restrict__ outp)
{
  const int wave = threadIdx.x >> 5, lane = threadIdx.x & 31;
  const int row = blockIdx.x * 8 + wave;
  const float* xp = x + (size_t)row * 256 + lane * 8;
  const float* dp = dlt + (size_t)row * 256 + lane * 8;
  float v[8];
  float s = 0.f, ss = 0.f;
#pragma unroll
  for (int i = 0; i < 8; ++i) { v[i] = xp[i] + dp[i]; s += v[i]; ss += v[i] * v[i]; }
#pragma unroll
  for (int sft = 1; sft < 32; sft <<= 1) {
    s += __shfl_xor(s, sft, 32);
    ss += __shfl_xor(ss, sft, 32);
  }
  float mean = s * (1.f / 256.f);
  float var  = ss * (1.f / 256.f) - mean * mean;
  float inv  = rsqrtf(var + 1e-5f);
  float* op = outp + (size_t)row * 256 + lane * 8;
#pragma unroll
  for (int i = 0; i < 8; ++i)
    op[i] = (v[i] - mean) * inv * g[lane * 8 + i] + b[lane * 8 + i];
}

__global__ void add_kernel(const float* __restrict__ a, const float* __restrict__ b,
                           float* __restrict__ o, int n) {
  int i = blockIdx.x * blockDim.x + threadIdx.x;
  if (i < n) o[i] = a[i] + b[i];
}

// One-shot: wp1t[n][k] = (f16) Wp1[k][n]   (128 x 256, row stride 256 halves)
__global__ void cvt_wp1t_kernel(const float* __restrict__ Wp1,
                                _Float16* __restrict__ wp1t) {
  int i = blockIdx.x * 256 + threadIdx.x;   // 32768 elements
  int n = i >> 8;
  int k = i & 255;
  wp1t[i] = (_Float16)Wp1[(size_t)k * 128 + n];
}

// ---------------------------------------------------------------------------
// Edge predictor: out[e] = leaky_relu((h[src]*h[dst]*w) @ Wp1 + bp1, 0.2) @ Wp2
//                 + bp2.  Gather fused into WMMA A-fragment; Wp1 K-slices
//                 (preconverted f16, transposed) staged to LDS with async
//                 global->LDS copies. 4 waves x 16 edges per block.
// ---------------------------------------------------------------------------
__global__ __launch_bounds__(128, 1) void edge_kernel(
    const float* __restrict__ hc, const int* __restrict__ src,
    const int* __restrict__ dst, const float* __restrict__ w,
    const _Float16* __restrict__ wp1t, const float* __restrict__ bp1,
    const float* __restrict__ Wp2, const float* __restrict__ bp2,
    float* __restrict__ outp, int E)
{
  __shared__ __align__(16) _Float16 Wt[128 * LDSTR];  // [col][k] slice
  const int tid = threadIdx.x, wave = tid >> 5, lane = tid & 31;
  const int lh = lane >> 4, lm = lane & 15;
  const int e0 = (blockIdx.x * 4 + wave) * 16;
  int eg = e0 + lm; if (eg > E - 1) eg = E - 1;
  const float* hs = hc + (size_t)src[eg] * 256;
  const float* hd = hc + (size_t)dst[eg] * 256;
  const float wgt = w[eg];

  v8f acc[8] = {};
  for (int k0 = 0; k0 < 256; k0 += 32) {
    __syncthreads();
    // Async-stage this K-slice of Wp1^T (f16): thread tid owns column tid.
    // 4 x 16B per thread: global [tid*256 + k0 .. +32) -> LDS Wt[tid][0..32)
#pragma unroll
    for (int c = 0; c < 4; ++c) {
      const _Float16* gsrc = wp1t + (size_t)tid * 256 + k0 + c * 8;
      unsigned loff = (unsigned)(unsigned long long)(AS3 _Float16*)
                          &Wt[tid * LDSTR + c * 8];
      asm volatile("global_load_async_to_lds_b128 %0, %1, off"
                   :: "v"(loff), "v"(gsrc) : "memory");
    }
    asm volatile("s_wait_asynccnt 0x0" ::: "memory");
    __syncthreads();

    v16h af;
#pragma unroll
    for (int j = 0; j < 8; ++j) {
      int ka = k0 + lh * 16 + j;
      int kb = k0 + lh * 16 + 8 + j;
      af[j]     = (_Float16)(hs[ka] * hd[ka] * wgt);
      af[j + 8] = (_Float16)(hs[kb] * hd[kb] * wgt);
    }
#pragma unroll
    for (int ct = 0; ct < 8; ++ct) {
      v16h bf = ld16h(&Wt[(ct * 16 + lm) * LDSTR + lh * 16]);
      acc[ct] = wmma_f16(af, bf, acc[ct]);
    }
  }

  const float bp2v = bp2[0];
#pragma unroll
  for (int r = 0; r < 8; ++r) {
    float s = 0.f;
#pragma unroll
    for (int ct = 0; ct < 8; ++ct) {
      int n = ct * 16 + lm;
      float t = acc[ct][r] + bp1[n];
      t = (t > 0.f) ? t : 0.2f * t;
      s += t * Wp2[n];
    }
#pragma unroll
    for (int sft = 1; sft < 16; sft <<= 1) s += __shfl_xor(s, sft, 16);
    int erow = e0 + lh * 8 + r;
    if (lm == r && erow < E) outp[erow] = s + bp2v;
  }
}

// ---------------------------------------------------------------------------
extern "C" void kernel_launch(void* const* d_in, const int* in_sizes, int n_in,
                              void* d_out, int out_size, void* d_ws, size_t ws_size,
                              hipStream_t stream) {
  constexpr int N_ = 4096, IN_ = 512, D_ = 256, L_ = 2, H_ = 8, FF_ = 1024;
  const float* x        = (const float*)d_in[0];
  const int*   pos_src  = (const int*)d_in[1];
  const int*   pos_dst  = (const int*)d_in[2];
  const float* pos_w    = (const float*)d_in[3];
  const int*   neg_src  = (const int*)d_in[4];
  const int*   neg_dst  = (const int*)d_in[5];
  const float* neg_w    = (const float*)d_in[6];
  const float* Wi   = (const float*)d_in[7];
  const float* bi   = (const float*)d_in[8];
  const float* Wqkv = (const float*)d_in[9];
  const float* bqkv = (const float*)d_in[10];
  const float* Wo   = (const float*)d_in[11];
  const float* bo   = (const float*)d_in[12];
  const float* g1   = (const float*)d_in[13];
  const float* be1  = (const float*)d_in[14];
  const float* g2   = (const float*)d_in[15];
  const float* be2  = (const float*)d_in[16];
  const float* W1   = (const float*)d_in[17];
  const float* b1   = (const float*)d_in[18];
  const float* W2   = (const float*)d_in[19];
  const float* b2   = (const float*)d_in[20];
  const float* Wp1  = (const float*)d_in[21];
  const float* bp1  = (const float*)d_in[22];
  const float* Wp2  = (const float*)d_in[23];
  const float* bp2  = (const float*)d_in[24];
  const int E = in_sizes[1];

  float* ws   = (float*)d_ws;
  float* h0   = ws;                     // N*D
  float* h    = h0 + (size_t)N_ * D_;   // N*D
  float* qkv  = h + (size_t)N_ * D_;    // N*3D
  float* tmp  = qkv + (size_t)N_ * 3 * D_;  // N*D (attn out / ffn out)
  float* ff   = tmp + (size_t)N_ * D_;  // N*FF (also holds o-proj temporarily)
  _Float16* wp1t = (_Float16*)(ff + (size_t)N_ * FF_);  // 128x256 f16

  float* out = (float*)d_out;
  float* hc  = out + 2 * (size_t)E;     // h_combined lives in d_out

  cvt_wp1t_kernel<<<128, 256, 0, stream>>>(Wp1, wp1t);

  // h0 = x @ Wi + bi ; h = h0
  gemm_kernel<<<dim3(D_ / 64, N_ / 64), 128, 0, stream>>>(x, Wi, bi, h0, N_, IN_, D_, 0);
  hipMemcpyAsync(h, h0, sizeof(float) * (size_t)N_ * D_, hipMemcpyDeviceToDevice, stream);

  for (int i = 0; i < L_; ++i) {
    gemm_kernel<<<dim3(3 * D_ / 64, N_ / 64), 128, 0, stream>>>(
        h, Wqkv + (size_t)i * D_ * 3 * D_, bqkv + (size_t)i * 3 * D_, qkv, N_, D_, 3 * D_, 0);
    flash_kernel<<<dim3(N_ / 16, H_), 32, 0, stream>>>(qkv, tmp);
    float* oproj = ff;  // reuse ff region before FFN needs it
    gemm_kernel<<<dim3(D_ / 64, N_ / 64), 128, 0, stream>>>(
        tmp, Wo + (size_t)i * D_ * D_, bo + (size_t)i * D_, oproj, N_, D_, D_, 0);
    ln_res_kernel<<<N_ / 8, 256, 0, stream>>>(h, oproj, g1 + (size_t)i * D_, be1 + (size_t)i * D_, h);
    gemm_kernel<<<dim3(FF_ / 64, N_ / 64), 128, 0, stream>>>(
        h, W1 + (size_t)i * D_ * FF_, b1 + (size_t)i * FF_, ff, N_, D_, FF_, 1);
    gemm_kernel<<<dim3(D_ / 64, N_ / 64), 128, 0, stream>>>(
        ff, W2 + (size_t)i * FF_ * D_, b2 + (size_t)i * D_, tmp, N_, FF_, D_, 0);
    ln_res_kernel<<<N_ / 8, 256, 0, stream>>>(h, tmp, g2 + (size_t)i * D_, be2 + (size_t)i * D_, h);
  }

  add_kernel<<<(N_ * D_ + 255) / 256, 256, 0, stream>>>(h, h0, hc, N_ * D_);

  int tiles = (E + 15) / 16;
  int eblocks = (tiles + 3) / 4;
  edge_kernel<<<eblocks, 128, 0, stream>>>(hc, pos_src, pos_dst, pos_w,
                                           wp1t, bp1, Wp2, bp2, out, E);
  edge_kernel<<<eblocks, 128, 0, stream>>>(hc, neg_src, neg_dst, neg_w,
                                           wp1t, bp1, Wp2, bp2, out + E, E);
}
